// FeatureAdaptMixer_11922829214479
// MI455X (gfx1250) — matrix operand order repaired and checked by reading, not compiled, not run on hardware
//
#include <hip/hip_runtime.h>

typedef _Float16 half_t;
typedef half_t v16h __attribute__((ext_vector_type(16)));
typedef half_t v8h  __attribute__((ext_vector_type(8)));
typedef float  v8f  __attribute__((ext_vector_type(8)));

namespace {

constexpr int BATCH = 4;
constexpr int CHN   = 256;
constexpr int HH    = 128;
constexpr int WW    = 128;
constexpr int HW    = HH * WW;      // 16384
constexpr int NHEAD = 16;
constexpr int HDIM  = 16;

// ------------------------- small utility kernels -------------------------

__global__ void k_cvt_f16(const float* __restrict__ s, half_t* __restrict__ d, int n) {
  int i = blockIdx.x * 256 + threadIdx.x;
  if (i < n) d[i] = (half_t)s[i];
}

__global__ void k_bn_prep(const float* __restrict__ g, const float* __restrict__ b,
                          const float* __restrict__ m, const float* __restrict__ v,
                          float* __restrict__ sc, float* __restrict__ bi) {
  int i = threadIdx.x;  // 256 channels
  float s = g[i] * rsqrtf(v[i] + 1e-5f);
  sc[i] = s;
  bi[i] = b[i] - m[i] * s;
}

// tbl[head][query(64)][key(64)]
__global__ void k_bias_tbl(const float* __restrict__ rel, float* __restrict__ tbl) {
  int i = blockIdx.x * 256 + threadIdx.x;  // 16*64*64
  int hd = i >> 12, nm = i & 4095, n = nm >> 6, m = nm & 63;
  int r0 = (n >> 3) - (m >> 3) + 7;
  int r1 = (n & 7) - (m & 7) + 7;
  tbl[i] = rel[(r0 * 15 + r1) * NHEAD + hd];
}

__global__ void k_add(const half_t* __restrict__ a, const half_t* __restrict__ b,
                      half_t* __restrict__ o, int n) {
  int i = blockIdx.x * 256 + threadIdx.x;
  if (i < n) o[i] = (half_t)((float)a[i] + (float)b[i]);
}

// ------------------------- implicit-GEMM WMMA conv -------------------------
// Out[b][oc][p] = sum_{c,kh,kw} W[oc][c*KH*KW+kh*KW+kw] * X[b][c][oh+kh-PAD][ow+kw-PAD]
// Block tile: 128 (M=oc) x 64 (N=spatial); 8 waves as 4(M) x 2(N); wave tile 32x32.
// Double-buffered LDS; B tile loaded N-major (coalesced; b128 fast path for 1x1).
template <int KH, int KW, int PAD, int CIN, bool BN, bool OUT_F32>
__global__ __launch_bounds__(256) void k_conv_wmma(
    const half_t* __restrict__ X, const half_t* __restrict__ Wt,
    const float* __restrict__ bsc, const float* __restrict__ bbi,
    half_t* __restrict__ Yh, float* __restrict__ Yf, int OC) {
  constexpr int KTAP = KH * KW;
  const int KTOT = CIN * KTAP;
  const int nsteps = KTOT / 32;
  const int b = blockIdx.z;
  const int ocBase = blockIdx.y * 128;
  const int nBase = blockIdx.x * 64;

  __shared__ half_t Al[2][128][40];  // [buf][m][k], stride 40 halfs (80B, 16B aligned)
  __shared__ half_t Bl[2][64][40];   // [buf][n][k]

  const int tid = threadIdx.x;
  const int lane = tid & 31;
  const int wv = tid >> 5;
  const int wm = wv >> 1, wn = wv & 1;
  const int m0 = wm * 32, n0 = wn * 32;
  const int r = lane & 15;
  const int kbA = (lane < 16) ? 0 : 16;   // A frag K base (low half)
  const int kbB = (lane < 16) ? 0 : 16;   // B frag K base

  const half_t* Xb = X + (size_t)b * CIN * HW;

  // per-thread invariant load mapping
  const int a_row = tid >> 1;            // A: 128 rows x (2 x 16 halfs)
  const int a_kc = (tid & 1) * 16;
  const int b_kk = tid >> 3;             // B: k lane 0..31
  const int b_ng = (tid & 7) * 8;        // B: 8 consecutive n
  const int p0 = nBase + b_ng;
  const int oh = p0 >> 7;                // whole 64-tile shares one row (W=128)
  const int ow0 = p0 & 127;

  auto load_tiles = [&](int k0, int buf) {
    // ---- A tile: weights, contiguous [OC][KTOT] ----
    const half_t* src = Wt + (size_t)(ocBase + a_row) * KTOT + k0 + a_kc;
    *(v8h*)&Al[buf][a_row][a_kc] = *(const v8h*)src;
    *(v8h*)&Al[buf][a_row][a_kc + 8] = *(const v8h*)(src + 8);
    if (k0 + 32 < KTOT) __builtin_prefetch(src + 32, 0, 1);
    // ---- B tile: im2col, loaded along N ----
    if (KTAP == 1) {
      // pure GEMM: one aligned 16B vector load per thread, fully coalesced
      const half_t* bs = Xb + (size_t)(k0 + b_kk) * HW + p0;
      v8h v = *(const v8h*)bs;
#pragma unroll
      for (int e = 0; e < 8; ++e) Bl[buf][b_ng + e][b_kk] = v[e];
    } else {
      int kg = k0 + b_kk;
      int c = kg / KTAP;
      int rem = kg - c * KTAP;
      int kh = rem / KW;
      int kw = rem - kh * KW;
      int ih = oh + kh - PAD;
      bool rowok = (unsigned)ih < (unsigned)HH;
      const half_t* rowp = Xb + (size_t)c * HW + ih * WW;
#pragma unroll
      for (int e = 0; e < 8; ++e) {
        int iw = ow0 + e + kw - PAD;
        half_t vv = (half_t)0.f;
        if (rowok && (unsigned)iw < (unsigned)WW) vv = rowp[iw];
        Bl[buf][b_ng + e][b_kk] = vv;
      }
    }
  };

  v8f acc[2][2] = {};

  load_tiles(0, 0);
  __syncthreads();

  for (int step = 0; step < nsteps; ++step) {
    const int buf = step & 1;

    // fragment loads from current buffer (issued before next-tile stores so the
    // WMMAs only wait on these ds_loads)
    v16h af[2], bf[2];
#pragma unroll
    for (int mi = 0; mi < 2; ++mi) {
      v8h lo = *(const v8h*)&Al[buf][m0 + mi * 16 + r][kbA];
      v8h hi = *(const v8h*)&Al[buf][m0 + mi * 16 + r][kbA + 8];
#pragma unroll
      for (int e = 0; e < 8; ++e) { af[mi][e] = lo[e]; af[mi][8 + e] = hi[e]; }
    }
#pragma unroll
    for (int ni = 0; ni < 2; ++ni) {
      v8h lo = *(const v8h*)&Bl[buf][n0 + ni * 16 + r][kbB];
      v8h hi = *(const v8h*)&Bl[buf][n0 + ni * 16 + r][kbB + 8];
#pragma unroll
      for (int e = 0; e < 8; ++e) { bf[ni][e] = lo[e]; bf[ni][8 + e] = hi[e]; }
    }

    if (step + 1 < nsteps) load_tiles((step + 1) * 32, buf ^ 1);

#pragma unroll
    for (int mi = 0; mi < 2; ++mi)
#pragma unroll
      for (int ni = 0; ni < 2; ++ni)
        acc[mi][ni] = __builtin_amdgcn_wmma_f32_16x16x32_f16(
            false, af[mi], false, bf[ni], (short)0, acc[mi][ni], false, false);

    __syncthreads();
  }

  // epilogue: fused BN, write f16 or f32
#pragma unroll
  for (int mi = 0; mi < 2; ++mi)
#pragma unroll
    for (int ni = 0; ni < 2; ++ni)
#pragma unroll
      for (int rr = 0; rr < 8; ++rr) {
        int m = ocBase + m0 + mi * 16 + ((lane < 16) ? rr : 8 + rr);
        int n = nBase + n0 + ni * 16 + r;
        float vv = acc[mi][ni][rr];
        if (BN) vv = vv * bsc[m] + bbi[m];
        size_t off = ((size_t)b * OC + m) * HW + n;
        if (OUT_F32) Yf[off] = vv; else Yh[off] = (half_t)vv;
      }
}

// ------------------------- windowed dual-dot attention -------------------------
// One block = one 8x8 window; 8 waves, each wave handles 2 heads sequentially.
// dots = [q|qh](64x32) @ [k|kh]^T  -> single K=32 WMMA per 16x16 tile.
__global__ __launch_bounds__(256) void k_attn(
    const half_t* __restrict__ QKV,  // [B][768][H][W] (q:0..255, k:256..511, v:512..767)
    const half_t* __restrict__ QHp,  // [B][256][H][W]
    const half_t* __restrict__ KHp,  // [B][256][H][W]
    const float* __restrict__ BT,    // [16][64][64]
    half_t* __restrict__ LOW) {      // [B][256][H][W]
  extern __shared__ half_t smem[];
  const int tid = threadIdx.x;
  const int lane = tid & 31;
  const int wv = tid >> 5;
  const int b = blockIdx.y;
  const int wy = blockIdx.x >> 4;
  const int wx = blockIdx.x & 15;
  const int ph0 = wy * 8, pw0 = wx * 8;

  // per-wave LDS slice: qcat[64][40] | kcat[64][40] | P[64][72]; V^T overlays qcat
  half_t* q_s = smem + (size_t)wv * 9728;
  half_t* k_s = q_s + 2560;
  half_t* p_s = k_s + 2560;
  half_t* v_s = q_s;

  const int r = lane & 15;
  const int kbA = (lane < 16) ? 0 : 16;
  const int kbB = (lane < 16) ? 0 : 16;

  for (int it = 0; it < 2; ++it) {
    const int hd = it * 8 + wv;
    const half_t* Qg = QKV + ((size_t)b * 768 + hd * HDIM) * HW;
    const half_t* Kg = QKV + ((size_t)b * 768 + 256 + hd * HDIM) * HW;
    const half_t* Vg = QKV + ((size_t)b * 768 + 512 + hd * HDIM) * HW;
    const half_t* Qh = QHp + ((size_t)b * CHN + hd * HDIM) * HW;
    const half_t* Kh = KHp + ((size_t)b * CHN + hd * HDIM) * HW;

    // stage concat matrices: [token][0:16]=attn part, [16:32]=high-freq part
#pragma unroll
    for (int i = 0; i < 32; ++i) {
      int lin = i * 32 + lane;  // 0..1023 = d(16) x token(64)
      int d = lin >> 6, tok = lin & 63;
      int go = d * HW + (ph0 + (tok >> 3)) * WW + pw0 + (tok & 7);
      q_s[tok * 40 + d] = Qg[go];
      q_s[tok * 40 + 16 + d] = Qh[go];
      k_s[tok * 40 + d] = Kg[go];
      k_s[tok * 40 + 16 + d] = Kh[go];
    }

    v16h af[4], bf[4];
#pragma unroll
    for (int mi = 0; mi < 4; ++mi) {
      v8h lo = *(const v8h*)&q_s[(mi * 16 + r) * 40 + kbA];
      v8h hi = *(const v8h*)&q_s[(mi * 16 + r) * 40 + kbA + 8];
#pragma unroll
      for (int e = 0; e < 8; ++e) { af[mi][e] = lo[e]; af[mi][8 + e] = hi[e]; }
    }
#pragma unroll
    for (int ni = 0; ni < 4; ++ni) {
      v8h lo = *(const v8h*)&k_s[(ni * 16 + r) * 40 + kbB];
      v8h hi = *(const v8h*)&k_s[(ni * 16 + r) * 40 + kbB + 8];
#pragma unroll
      for (int e = 0; e < 8; ++e) { bf[ni][e] = lo[e]; bf[ni][8 + e] = hi[e]; }
    }
    v8f acc[4][4];
#pragma unroll
    for (int mi = 0; mi < 4; ++mi)
#pragma unroll
      for (int ni = 0; ni < 4; ++ni) {
        v8f z = {};
        acc[mi][ni] = __builtin_amdgcn_wmma_f32_16x16x32_f16(
            false, af[mi], false, bf[ni], (short)0, z, false, false);
      }

    // scale (HD^-0.5 = 0.25) and relative position bias
    const float* bt = BT + hd * 4096;
#pragma unroll
    for (int mi = 0; mi < 4; ++mi)
#pragma unroll
      for (int ni = 0; ni < 4; ++ni)
#pragma unroll
        for (int rr = 0; rr < 8; ++rr) {
          int mrow = mi * 16 + ((lane < 16) ? rr : 8 + rr);
          acc[mi][ni][rr] = acc[mi][ni][rr] * 0.25f + bt[mrow * 64 + ni * 16 + r];
        }

    // softmax over key axis: combine 4 N-tiles then 16-lane xor reduction
#pragma unroll
    for (int mi = 0; mi < 4; ++mi)
#pragma unroll
      for (int rr = 0; rr < 8; ++rr) {
        float mx = -1e30f;
#pragma unroll
        for (int ni = 0; ni < 4; ++ni) mx = fmaxf(mx, acc[mi][ni][rr]);
#pragma unroll
        for (int s = 1; s < 16; s <<= 1) mx = fmaxf(mx, __shfl_xor(mx, s, 32));
        float sum = 0.f;
#pragma unroll
        for (int ni = 0; ni < 4; ++ni) {
          float e = __expf(acc[mi][ni][rr] - mx);
          acc[mi][ni][rr] = e;
          sum += e;
        }
#pragma unroll
        for (int s = 1; s < 16; s <<= 1) sum += __shfl_xor(sum, s, 32);
        float inv = 1.f / sum;
#pragma unroll
        for (int ni = 0; ni < 4; ++ni) acc[mi][ni][rr] *= inv;
      }

    // P -> LDS (A-operand layout source), stage V^T (overlay q_s region)
#pragma unroll
    for (int mi = 0; mi < 4; ++mi)
#pragma unroll
      for (int ni = 0; ni < 4; ++ni)
#pragma unroll
        for (int rr = 0; rr < 8; ++rr) {
          int mrow = mi * 16 + ((lane < 16) ? rr : 8 + rr);
          p_s[mrow * 72 + ni * 16 + r] = (half_t)acc[mi][ni][rr];
        }
#pragma unroll
    for (int i = 0; i < 32; ++i) {
      int lin = i * 32 + lane;
      int d = lin >> 6, tok = lin & 63;
      v_s[d * 72 + tok] = Vg[d * HW + (ph0 + (tok >> 3)) * WW + pw0 + (tok & 7)];
    }

    // out = P(64x64) @ V(64x16): K=64 as two chained K=32 WMMAs
    v8f o[4] = {};
#pragma unroll
    for (int kk = 0; kk < 2; ++kk) {
      v16h vb;
      {
        int kb = kk * 32 + kbB;
        v8h lo = *(const v8h*)&v_s[r * 72 + kb];
        v8h hi = *(const v8h*)&v_s[r * 72 + kb + 8];
#pragma unroll
        for (int e = 0; e < 8; ++e) { vb[e] = lo[e]; vb[8 + e] = hi[e]; }
      }
#pragma unroll
      for (int mi = 0; mi < 4; ++mi) {
        int kb = kk * 32 + kbA;
        v8h lo = *(const v8h*)&p_s[(mi * 16 + r) * 72 + kb];
        v8h hi = *(const v8h*)&p_s[(mi * 16 + r) * 72 + kb + 8];
        v16h pa;
#pragma unroll
        for (int e = 0; e < 8; ++e) { pa[e] = lo[e]; pa[8 + e] = hi[e]; }
        o[mi] = __builtin_amdgcn_wmma_f32_16x16x32_f16(
            false, pa, false, vb, (short)0, o[mi], false, false);
      }
    }

    half_t* Lg = LOW + ((size_t)b * CHN + hd * HDIM) * HW;
#pragma unroll
    for (int mi = 0; mi < 4; ++mi)
#pragma unroll
      for (int rr = 0; rr < 8; ++rr) {
        int tok = mi * 16 + ((lane < 16) ? rr : 8 + rr);
        Lg[r * HW + (ph0 + (tok >> 3)) * WW + pw0 + (tok & 7)] = (half_t)o[mi][rr];
      }
  }
}

// ------------------------- gating / depthwise stages -------------------------

__global__ void k_pool(const half_t* __restrict__ low, const half_t* __restrict__ high,
                       float* __restrict__ pooled) {
  int p = blockIdx.x * 256 + threadIdx.x;  // < HW
  int b = blockIdx.y;
  float s = 0.f, mx = -1e30f;
  for (int c = 0; c < CHN; ++c) {
    size_t off = ((size_t)b * CHN + c) * HW + p;
    float v = (float)low[off] + (float)high[off];
    s += v;
    mx = fmaxf(mx, v);
  }
  pooled[((size_t)b * 2 + 0) * HW + p] = s * (1.f / CHN);
  pooled[((size_t)b * 2 + 1) * HW + p] = mx;
}

__global__ void k_hybrid(const float* __restrict__ pooled, const float* __restrict__ w,
                         const float* __restrict__ bias, float* __restrict__ sig) {
  int p = blockIdx.x * 256 + threadIdx.x;
  int b = blockIdx.y;
  int oh = p >> 7, ow = p & 127;
  for (int oc = 0; oc < 2; ++oc) {
    float a = bias[oc];
    for (int ic = 0; ic < 2; ++ic)
      for (int kh = 0; kh < 7; ++kh) {
        int ih = oh + kh - 3;
        if ((unsigned)ih >= (unsigned)HH) continue;
        for (int kw = 0; kw < 7; ++kw) {
          int iw = ow + kw - 3;
          if ((unsigned)iw >= (unsigned)WW) continue;
          a += w[((oc * 2 + ic) * 7 + kh) * 7 + kw] *
               pooled[((size_t)b * 2 + ic) * HW + ih * WW + iw];
        }
      }
    sig[((size_t)b * 2 + oc) * HW + p] = 1.f / (1.f + __expf(-a));
  }
}

__global__ void k_gate(const half_t* __restrict__ low, const half_t* __restrict__ high,
                       const float* __restrict__ sig, half_t* __restrict__ out) {
  size_t i = (size_t)blockIdx.x * 256 + threadIdx.x;  // B*C*HW
  int b = (int)(i >> 22);        // C*HW = 2^22
  int p = (int)(i & (HW - 1));
  float s0 = sig[((size_t)b * 2 + 0) * HW + p];
  float s1 = sig[((size_t)b * 2 + 1) * HW + p];
  out[i] = (half_t)((float)low[i] * s0 + (float)high[i] * s1);
}

__global__ void k_dwconv(const half_t* __restrict__ in, const float* __restrict__ w,
                         const float* __restrict__ bsc, const float* __restrict__ bbi,
                         half_t* __restrict__ out) {
  size_t i = (size_t)blockIdx.x * 256 + threadIdx.x;  // B*C*HW
  int p = (int)(i & (HW - 1));
  int c = (int)((i >> 14) & (CHN - 1));
  int oh = p >> 7, ow = p & 127;
  size_t base = i - p;
  float a = 0.f;
  for (int kh = 0; kh < 7; ++kh) {
    int ih = oh + kh - 3;
    if ((unsigned)ih >= (unsigned)HH) continue;
    for (int kw = 0; kw < 7; ++kw) {
      int iw = ow + kw - 3;
      if ((unsigned)iw >= (unsigned)WW) continue;
      a += w[c * 49 + kh * 7 + kw] * (float)in[base + ih * WW + iw];
    }
  }
  out[i] = (half_t)(a * bsc[c] + bbi[c]);
}

}  // namespace

// ------------------------- orchestration -------------------------

extern "C" void kernel_launch(void* const* d_in, const int* in_sizes, int n_in,
                              void* d_out, int out_size, void* d_ws, size_t ws_size,
                              hipStream_t stream) {
  const float* x     = (const float*)d_in[0];
  const float* qkv_w = (const float*)d_in[1];
  const float* l1_w  = (const float*)d_in[2];
  const float* l1_g  = (const float*)d_in[3];
  const float* l1_b  = (const float*)d_in[4];
  const float* l1_m  = (const float*)d_in[5];
  const float* l1_v  = (const float*)d_in[6];
  const float* l2_w  = (const float*)d_in[7];
  const float* l2_g  = (const float*)d_in[8];
  const float* l2_b  = (const float*)d_in[9];
  const float* l2_m  = (const float*)d_in[10];
  const float* l2_v  = (const float*)d_in[11];
  const float* rel   = (const float*)d_in[12];
  const float* hyb_w = (const float*)d_in[13];
  const float* hyb_b = (const float*)d_in[14];
  const float* sm_w  = (const float*)d_in[15];
  const float* sm_g  = (const float*)d_in[16];
  const float* sm_b  = (const float*)d_in[17];
  const float* sm_m  = (const float*)d_in[18];
  const float* sm_v  = (const float*)d_in[19];
  const float* dw_w  = (const float*)d_in[20];
  const float* pj_g  = (const float*)d_in[21];
  const float* pj_b  = (const float*)d_in[22];
  const float* pj_m  = (const float*)d_in[23];
  const float* pj_v  = (const float*)d_in[24];
  const float* pw_w  = (const float*)d_in[25];

  char* base = (char*)d_ws;
  size_t off = 0;
  auto alloc = [&](size_t bytes) -> char* {
    char* p = base + off;
    off = (off + bytes + 255) & ~(size_t)255;
    return p;
  };

  const size_t tens = (size_t)BATCH * CHN * HW;  // 16,777,216 elements

  half_t* wq  = (half_t*)alloc((size_t)768 * 256 * 2);
  half_t* wl1 = (half_t*)alloc((size_t)256 * 2304 * 2);
  half_t* wl2 = (half_t*)alloc((size_t)256 * 256 * 2);
  half_t* wsm = (half_t*)alloc((size_t)256 * 6400 * 2);
  half_t* wpw = (half_t*)alloc((size_t)256 * 256 * 2);
  float* sc1 = (float*)alloc(1024); float* bi1 = (float*)alloc(1024);
  float* sc2 = (float*)alloc(1024); float* bi2 = (float*)alloc(1024);
  float* scs = (float*)alloc(1024); float* bis = (float*)alloc(1024);
  float* scp = (float*)alloc(1024); float* bip = (float*)alloc(1024);
  float* btbl = (float*)alloc((size_t)16 * 64 * 64 * 4);
  float* pooled = (float*)alloc((size_t)BATCH * 2 * HW * 4);
  float* sig    = (float*)alloc((size_t)BATCH * 2 * HW * 4);
  half_t* xh   = (half_t*)alloc(tens * 2);
  half_t* qh   = (half_t*)alloc(tens * 2);
  half_t* kh   = (half_t*)alloc(tens * 2);
  half_t* high = (half_t*)alloc(tens * 2);
  half_t* low  = (half_t*)alloc(tens * 2);
  half_t* qkvo = (half_t*)alloc(tens * 3 * 2);
  half_t* gated = xh;           // x no longer needed after the three input convs
  half_t* smout = qkvo;         // qkv no longer needed after attention
  half_t* dwout = qkvo + tens;

  auto cvt = [&](const float* s, half_t* d, int n) {
    k_cvt_f16<<<(n + 255) / 256, 256, 0, stream>>>(s, d, n);
  };

  // prep: weight/bn/bias tables
  cvt(qkv_w, wq, 768 * 256);
  cvt(l1_w, wl1, 256 * 2304);
  cvt(l2_w, wl2, 256 * 256);
  cvt(sm_w, wsm, 256 * 6400);
  cvt(pw_w, wpw, 256 * 256);
  k_bn_prep<<<1, 256, 0, stream>>>(l1_g, l1_b, l1_m, l1_v, sc1, bi1);
  k_bn_prep<<<1, 256, 0, stream>>>(l2_g, l2_b, l2_m, l2_v, sc2, bi2);
  k_bn_prep<<<1, 256, 0, stream>>>(sm_g, sm_b, sm_m, sm_v, scs, bis);
  k_bn_prep<<<1, 256, 0, stream>>>(pj_g, pj_b, pj_m, pj_v, scp, bip);
  k_bias_tbl<<<256, 256, 0, stream>>>(rel, btbl);
  cvt(x, xh, (int)tens);

  // dense WMMA convs
  dim3 g768(HW / 64, 768 / 128, BATCH);
  dim3 g256(HW / 64, 256 / 128, BATCH);
  k_conv_wmma<1, 1, 0, 256, false, false><<<g768, 256, 0, stream>>>(
      xh, wq, nullptr, nullptr, qkvo, nullptr, 768);
  k_conv_wmma<3, 3, 1, 256, true, false><<<g256, 256, 0, stream>>>(
      xh, wl1, sc1, bi1, qh, nullptr, 256);
  k_conv_wmma<1, 1, 0, 256, true, false><<<g256, 256, 0, stream>>>(
      xh, wl2, sc2, bi2, kh, nullptr, 256);
  k_add<<<(int)((tens + 255) / 256), 256, 0, stream>>>(qh, kh, high, (int)tens);

  // windowed attention (dual dot product folded into K=32)
  k_attn<<<dim3(256, BATCH), 256, 155648, stream>>>(qkvo, qh, kh, btbl, low);

  // hybrid gate
  k_pool<<<dim3(HW / 256, BATCH), 256, 0, stream>>>(low, high, pooled);
  k_hybrid<<<dim3(HW / 256, BATCH), 256, 0, stream>>>(pooled, hyb_w, hyb_b, sig);
  k_gate<<<(int)((tens + 255) / 256), 256, 0, stream>>>(low, high, sig, gated);

  // smooth 5x5 + BN, depthwise 7x7 + BN, final pointwise to fp32 output
  k_conv_wmma<5, 5, 2, 256, true, false><<<g256, 256, 0, stream>>>(
      gated, wsm, scs, bis, smout, nullptr, 256);
  k_dwconv<<<(int)((tens + 255) / 256), 256, 0, stream>>>(smout, dw_w, scp, bip, dwout);
  k_conv_wmma<1, 1, 0, 256, false, true><<<g256, 256, 0, stream>>>(
      dwout, wpw, nullptr, nullptr, nullptr, (float*)d_out, 256);
}
